// Model_2_7026566496750
// MI455X (gfx1250) — matrix operand order repaired and checked by reading, not compiled, not run on hardware
//
#include <hip/hip_runtime.h>
#include <hip/hip_bf16.h>
#include <math.h>

#define B_   64
#define N_   750
#define F_   8
#define NF_  250
#define K3_  16000   // 64 channels * 250 positions (real K)
#define K3P_ 16128   // padded K = 63 * 256, so 8 waves x 63 const-trip iterations

typedef _Float16 v16h __attribute__((ext_vector_type(16)));
typedef float    v8f  __attribute__((ext_vector_type(8)));

__device__ __forceinline__ float bn_relu(float x, float g, float b, float inv) {
    float y = x * inv * g + b;
    return y > 0.0f ? y : 0.0f;
}

// ---------------- 1) s = relu(BN(x_sample @ W1 + b1)) -> xcat[:, 0:250] -------------
__global__ void k_linear1(const float* __restrict__ xs, const float* __restrict__ W1,
                          const float* __restrict__ b1, const float* __restrict__ gl,
                          const float* __restrict__ bl, float* __restrict__ xcat) {
    __shared__ float xrow[NF_];
    int b = blockIdx.x, tid = threadIdx.x;
    for (int k = tid; k < NF_; k += blockDim.x) xrow[k] = xs[b * NF_ + k];
    __syncthreads();
    const float inv = rsqrtf(1.0f + 1e-5f);
    for (int j = tid; j < 250; j += blockDim.x) {
        float acc = b1[j];
        for (int k = 0; k < NF_; ++k) acc += xrow[k] * W1[k * 250 + j];
        xcat[b * 1000 + j] = bn_relu(acc, gl[j], bl[j], inv);
    }
}

// ---------------- 2a) Xw = x_graph @ Wg  [B,N,8] ------------------------------------
__global__ void k_xw(const float* __restrict__ xg, const float* __restrict__ Wg,
                     float* __restrict__ Xw) {
    int idx = blockIdx.x * blockDim.x + threadIdx.x;
    if (idx >= B_ * N_) return;
    const float* r = xg + idx * F_;
    float x[F_];
    for (int f = 0; f < F_; ++f) x[f] = r[f];
    for (int f = 0; f < F_; ++f) {
        float acc = 0.0f;
        for (int k = 0; k < F_; ++k) acc += x[k] * Wg[k * F_ + f];
        Xw[idx * F_ + f] = acc;
    }
}

// ---------------- 2b) dn = rsqrt(deg); deg = #{j: (d_i-d_j)^2 < 0.01} ---------------
__global__ void k_deg(const float* __restrict__ xg, float* __restrict__ dn) {
    int node = blockIdx.x;           // b*N + i
    int b = node / N_;
    int tid = threadIdx.x;
    float di = xg[node * F_];
    const float* base = xg + b * N_ * F_;
    int cnt = 0;
    for (int j = tid; j < N_; j += blockDim.x) {
        float df = di - base[j * F_];
        cnt += (df * df < 0.01f) ? 1 : 0;   // includes j==i (self loop)
    }
    __shared__ int red[256];
    red[tid] = cnt;
    __syncthreads();
    for (int s = 128; s > 0; s >>= 1) {
        if (tid < s) red[tid] += red[tid + s];
        __syncthreads();
    }
    if (tid == 0) dn[node] = rsqrtf((float)red[0]);
}

// ---------------- 2c) g[b,i] = mean_f relu(BN1(dn_i * sum_j A_ij dn_j Xw[j] + bg)) --
__global__ void k_gcn(const float* __restrict__ xg, const float* __restrict__ Xw,
                      const float* __restrict__ dn, const float* __restrict__ bg,
                      const float* __restrict__ g1, const float* __restrict__ be1,
                      float* __restrict__ xcat) {
    int node = blockIdx.x;
    int b = node / N_, i = node % N_;
    int tid = threadIdx.x;
    float di = xg[node * F_];
    const float* dbase = xg + b * N_ * F_;
    const float* xwb   = Xw + b * N_ * F_;
    const float* dnb   = dn + b * N_;
    float acc[F_];
    for (int f = 0; f < F_; ++f) acc[f] = 0.0f;
    for (int j = tid; j < N_; j += blockDim.x) {
        float df = di - dbase[j * F_];
        if (df * df < 0.01f) {                // A == mask elementwise
            float w = dnb[j];
            const float* xr = xwb + j * F_;
            for (int f = 0; f < F_; ++f) acc[f] += w * xr[f];
        }
    }
    __shared__ float red[F_ * 256];
    for (int f = 0; f < F_; ++f) red[f * 256 + tid] = acc[f];
    __syncthreads();
    for (int s = 128; s > 0; s >>= 1) {
        if (tid < s)
            for (int f = 0; f < F_; ++f) red[f * 256 + tid] += red[f * 256 + tid + s];
        __syncthreads();
    }
    if (tid == 0) {
        const float inv = rsqrtf(1.0f + 1e-5f);
        float dni = dnb[i], g1v = g1[0], be1v = be1[0], gsum = 0.0f;
        for (int f = 0; f < F_; ++f) {
            float h = dni * red[f * 256] + bg[f];
            gsum += bn_relu(h, g1v, be1v, inv);
        }
        xcat[b * 1000 + 250 + i] = gsum * (1.0f / F_);
    }
}

// ---------------- 3) conv1(K=5,pad2) + BN + ReLU + maxpool2 -> [64,32,500] ----------
__global__ void k_conv1pool(const float* __restrict__ xcat, const float* __restrict__ Wc1,
                            const float* __restrict__ bc1, const float* __restrict__ gc1,
                            const float* __restrict__ bec1, float* __restrict__ pool1) {
    int idx = blockIdx.x * blockDim.x + threadIdx.x;    // b*32*500 + c*500 + p
    if (idx >= 64 * 32 * 500) return;
    int p = idx % 500, c = (idx / 500) % 32, b = idx / (500 * 32);
    const float* xr = xcat + b * 1000;
    float wk[5];
    for (int k = 0; k < 5; ++k) wk[k] = Wc1[c * 5 + k];
    const float inv = rsqrtf(1.0f + 1e-5f);
    float best = -3.0e38f;
    for (int t = 2 * p; t <= 2 * p + 1; ++t) {
        float acc = bc1[c];
        for (int k = 0; k < 5; ++k) {
            int l = t + k - 2;
            if ((unsigned)l < 1000u) acc += xr[l] * wk[k];
        }
        best = fmaxf(best, bn_relu(acc, gc1[c], bec1[c], inv));
    }
    pool1[idx] = best;
}

// ---------------- 4) conv2(K=5,pad2) + BN + ReLU + maxpool2 -> f16 A[64,K3P_] -------
__global__ void k_conv2pool(const float* __restrict__ pool1, const float* __restrict__ Wc2,
                            const float* __restrict__ bc2, const float* __restrict__ gc2,
                            const float* __restrict__ bec2, _Float16* __restrict__ a_h) {
    int idx = blockIdx.x * blockDim.x + threadIdx.x;    // b*64*250 + c*250 + p
    if (idx >= 64 * 64 * 250) return;
    int p = idx % 250, c = (idx / 250) % 64, b = idx / (250 * 64);
    const float* inb = pool1 + b * 32 * 500;
    const float* wr  = Wc2 + c * 160;
    const float inv = rsqrtf(1.0f + 1e-5f);
    float best = -3.0e38f;
    for (int t = 2 * p; t <= 2 * p + 1; ++t) {
        float acc = bc2[c];
        for (int ci = 0; ci < 32; ++ci) {
            const float* ir = inb + ci * 500;
            const float* w5 = wr + ci * 5;
            for (int k = 0; k < 5; ++k) {
                int l = t + k - 2;
                if ((unsigned)l < 500u) acc += ir[l] * w5[k];
            }
        }
        best = fmaxf(best, bn_relu(acc, gc2[c], bec2[c], inv));
    }
    a_h[b * K3P_ + c * 250 + p] = (_Float16)best;       // A-matrix, row-major [64,K3P_]
}

// ---------------- 4b) zero the K-pad tail of A rows ([16000,16128)) ----------------
__global__ void k_apad(_Float16* __restrict__ a_h) {
    int idx = blockIdx.x * blockDim.x + threadIdx.x;    // 64 rows * 128 pad cols
    if (idx >= 64 * (K3P_ - K3_)) return;
    int row = idx / (K3P_ - K3_), k = K3_ + idx % (K3P_ - K3_);
    a_h[row * K3P_ + k] = (_Float16)0.0f;
}

// ---------------- 5) Wc3 f32 [128,16000] -> f16 [128,K3P_] row-major, zero-padded ---
__global__ void k_cvt(const float* __restrict__ w, _Float16* __restrict__ o) {
    int idx = blockIdx.x * blockDim.x + threadIdx.x;    // 128 * K3P_
    if (idx >= 128 * K3P_) return;
    int row = idx / K3P_, k = idx % K3P_;
    o[idx] = (k < K3_) ? (_Float16)w[row * K3_ + k] : (_Float16)0.0f;
}

// ---------------- 6) conv3 as GEMM: C[64,128] = A[64,K] @ Bt[128,K]^T ---------------
// 8 waves per block, one 16x16 C tile per block (grid = 32 tiles => 256 waves).
// Each wave does a CONSTANT 63 iterations of its interleaved K-slice (K padded to
// 63*256) with v_wmma_f32_16x16x32_f16, unroll 7 restores load/wmma pipelining; the
// 8 partial v8f accumulators are then reduced through LDS. Fragments per ISA 7.12.2.
__global__ void k_conv3_wmma(const _Float16* __restrict__ A, const _Float16* __restrict__ Bt,
                             const float* __restrict__ bc3, float* __restrict__ x3) {
    int tid   = threadIdx.x;          // 0..255 (8 waves)
    int wave  = tid >> 5;             // 0..7  K-slice id
    int lane  = tid & 31;             // 0..31, EXEC all-ones per wave (uniform flow)
    int mTile = blockIdx.x & 3;       // 4 M-tiles (batch)
    int nTile = blockIdx.x >> 2;      // 8 N-tiles (out channels)
    int half  = lane >> 4;            // lane group 0/1
    int lmod  = lane & 15;
    const _Float16* aRow = A  + (long)(mTile * 16 + lmod) * K3P_ + wave * 32 + half * 8;
    const _Float16* bRow = Bt + (long)(nTile * 16 + lmod) * K3P_ + wave * 32 + half * 8;
    v8f c = {};
    union P { uint4 u[2]; v16h v; };
#pragma unroll 7
    for (int it = 0; it < 63; ++it) {                   // k0 = wave*32 + it*256
        const _Float16* ap = aRow + it * 256;
        const _Float16* bp = bRow + it * 256;
        P a, bm;
        // A lane fragment: K = k0 + {half*8..+7, 16+half*8..+7}
        a.u[0]  = *(const uint4*)(ap);
        a.u[1]  = *(const uint4*)(ap + 16);
        // B lane fragment: K = k0 + {half*8..+7, 16+half*8..+7}
        bm.u[0] = *(const uint4*)(bp);
        bm.u[1] = *(const uint4*)(bp + 16);
        c = __builtin_amdgcn_wmma_f32_16x16x32_f16(
                /*neg_a=*/false, a.v, /*neg_b=*/false, bm.v,
                /*c_mod=*/(short)0, c, /*reuse_a=*/false, /*reuse_b=*/false);
    }
    // cross-wave reduction of the 8 partial accumulators through LDS
    __shared__ float red[8 * 256];
    for (int v = 0; v < 8; ++v) red[wave * 256 + lane * 8 + v] = c[v];
    __syncthreads();
    // each thread owns one (lane,v) slot of the 16x16 tile
    int slane = tid >> 3;             // 0..31
    int sv    = tid & 7;              // 0..7
    float sum = 0.0f;
    for (int w = 0; w < 8; ++w) sum += red[w * 256 + slane * 8 + sv];
    int nCol = nTile * 16 + (slane & 15);                 // N = lane%16
    int m    = mTile * 16 + sv + 8 * (slane >> 4);        // M = v + 8*(lane/16)
    x3[m * 128 + nCol] = sum + bc3[nCol];
}

// ---------------- 7) FC -> BN/ReLU -> FC -> log_softmax ----------------------------
__global__ void k_head(const float* __restrict__ x3, const float* __restrict__ Wf,
                       const float* __restrict__ bf, const float* __restrict__ gf,
                       const float* __restrict__ bef, const float* __restrict__ Wo,
                       const float* __restrict__ bo, float* __restrict__ out) {
    __shared__ float xr[128];
    __shared__ float x4[64];
    __shared__ float z[12];
    int b = blockIdx.x, tid = threadIdx.x;          // 64 threads
    xr[tid]      = x3[b * 128 + tid];
    xr[tid + 64] = x3[b * 128 + 64 + tid];
    __syncthreads();
    const float inv = rsqrtf(1.0f + 1e-5f);
    float acc = bf[tid];
    for (int k = 0; k < 128; ++k) acc += xr[k] * Wf[k * 64 + tid];
    x4[tid] = bn_relu(acc, gf[tid], bef[tid], inv);
    __syncthreads();
    if (tid < 12) {
        float zz = bo[tid];
        for (int k = 0; k < 64; ++k) zz += x4[k] * Wo[k * 12 + tid];
        z[tid] = zz;
    }
    __syncthreads();
    if (tid < 12) {
        float m = z[0];
        for (int k = 1; k < 12; ++k) m = fmaxf(m, z[k]);
        float ssum = 0.0f;
        for (int k = 0; k < 12; ++k) ssum += __expf(z[k] - m);
        out[b * 12 + tid] = z[tid] - m - __logf(ssum);
    }
}

extern "C" void kernel_launch(void* const* d_in, const int* in_sizes, int n_in,
                              void* d_out, int out_size, void* d_ws, size_t ws_size,
                              hipStream_t stream) {
    const float* xs  = (const float*)d_in[0];   // [64,250]
    const float* xg  = (const float*)d_in[1];   // [64,750,8]
    const float* W1  = (const float*)d_in[2];
    const float* b1  = (const float*)d_in[3];
    const float* gl  = (const float*)d_in[4];
    const float* bl  = (const float*)d_in[5];
    const float* Wg  = (const float*)d_in[6];
    const float* bg  = (const float*)d_in[7];
    const float* g1  = (const float*)d_in[8];
    const float* be1 = (const float*)d_in[9];
    const float* Wc1 = (const float*)d_in[10];
    const float* bc1 = (const float*)d_in[11];
    const float* gc1 = (const float*)d_in[12];
    const float* bec1= (const float*)d_in[13];
    const float* Wc2 = (const float*)d_in[14];
    const float* bc2 = (const float*)d_in[15];
    const float* gc2 = (const float*)d_in[16];
    const float* bec2= (const float*)d_in[17];
    const float* Wc3 = (const float*)d_in[18];
    const float* bc3 = (const float*)d_in[19];
    const float* Wf  = (const float*)d_in[20];
    const float* bf  = (const float*)d_in[21];
    const float* gf  = (const float*)d_in[22];
    const float* bef = (const float*)d_in[23];
    const float* Wo  = (const float*)d_in[24];
    const float* bo  = (const float*)d_in[25];
    float* out = (float*)d_out;

    // workspace layout (bytes, 256-aligned)
    char* ws = (char*)d_ws;
    float*    xcat  = (float*)(ws + 0);          //  64*1000 f32   = 256000
    float*    dn    = (float*)(ws + 256000);     //  48000   f32   = 192000
    float*    Xw    = (float*)(ws + 448000);     //  384000  f32   = 1536000
    float*    pool1 = (float*)(ws + 1984000);    //  64*32*500 f32 = 4096000
    _Float16* a_h   = (_Float16*)(ws + 6080000); //  64*16128  f16 = 2064384
    _Float16* w3_h  = (_Float16*)(ws + 8144384); //  128*16128 f16 = 4128768
    float*    x3    = (float*)(ws + 12273152);   //  64*128 f32

    k_linear1   <<<64,    256, 0, stream>>>(xs, W1, b1, gl, bl, xcat);
    k_xw        <<<188,   256, 0, stream>>>(xg, Wg, Xw);
    k_deg       <<<48000, 256, 0, stream>>>(xg, dn);
    k_gcn       <<<48000, 256, 0, stream>>>(xg, Xw, dn, bg, g1, be1, xcat);
    k_conv1pool <<<4000,  256, 0, stream>>>(xcat, Wc1, bc1, gc1, bec1, pool1);
    k_conv2pool <<<4000,  256, 0, stream>>>(pool1, Wc2, bc2, gc2, bec2, a_h);
    k_apad      <<<32,    256, 0, stream>>>(a_h);
    k_cvt       <<<8064,  256, 0, stream>>>(Wc3, w3_h);
    k_conv3_wmma<<<32,    256, 0, stream>>>(a_h, w3_h, bc3, x3);
    k_head      <<<64,     64, 0, stream>>>(x3, Wf, bf, gf, bef, Wo, bo, out);
}
